// QuantLinear_59322088292778
// MI455X (gfx1250) — 8-bit matrix operand order repaired and checked by reading, compiled, not run on hardware
//
#include <hip/hip_runtime.h>
#include <hip/hip_bf16.h>
#include <stdint.h>

typedef __attribute__((ext_vector_type(8))) int v8i;

#define IN_F   4096
#define OUT_F  4096
#define GROUP  128
#define KTILES (IN_F / 64)
#define NTILES (OUT_F / 16)

__device__ __forceinline__ int q8clamp(float f) {
  int i = (int)rintf(f);
  i = i < -128 ? -128 : i;
  i = i >  127 ?  127 : i;
  return i;
}

__device__ __forceinline__ v8i load_frag(const int8_t* p) {
  const uint4 a = ((const uint4*)p)[0];
  const uint4 b = ((const uint4*)p)[1];
  return (v8i){(int)a.x, (int)a.y, (int)a.z, (int)a.w,
               (int)b.x, (int)b.y, (int)b.z, (int)b.w};
}

__global__ void quant_act_kernel(const float* __restrict__ x,
                                 int8_t* __restrict__ packedA,
                                 float* __restrict__ xscale) {
  __shared__ float red[256];
  const int row = blockIdx.x;
  const int tid = threadIdx.x;
  const float* xr = x + (size_t)row * IN_F;

  float4 v[4];
  float m = 0.f;
#pragma unroll
  for (int i = 0; i < 4; ++i) {
    v[i] = ((const float4*)xr)[tid * 4 + i];
    m = fmaxf(m, fmaxf(fmaxf(fabsf(v[i].x), fabsf(v[i].y)),
                       fmaxf(fabsf(v[i].z), fabsf(v[i].w))));
  }
  red[tid] = m;
  __syncthreads();
#pragma unroll
  for (int s = 128; s > 0; s >>= 1) {
    if (tid < s) red[tid] = fmaxf(red[tid], red[tid + s]);
    __syncthreads();
  }
  const float amax = fmaxf(red[0], 1e-30f);
  const float inv  = 127.0f / amax;
  if (tid == 0) xscale[row] = amax * (1.0f / 127.0f);

  uint32_t packed[4];
#pragma unroll
  for (int i = 0; i < 4; ++i) {
    const int a = q8clamp(v[i].x * inv);
    const int b = q8clamp(v[i].y * inv);
    const int c = q8clamp(v[i].z * inv);
    const int d = q8clamp(v[i].w * inv);
    packed[i] = (uint32_t)(a & 255) | ((uint32_t)(b & 255) << 8) |
                ((uint32_t)(c & 255) << 16) | ((uint32_t)(d & 255) << 24);
  }

  const int kt    = tid >> 2;
  const int pos   = tid & 3;
  const int mtile = row >> 4;
  const int lane_lo = row & 15;
  const size_t base = ((size_t)mtile * KTILES + kt) * 32;
  *(uint2*)(packedA + (base + lane_lo) * 32 + pos * 8) =
      make_uint2(packed[0], packed[1]);
  *(uint2*)(packedA + (base + lane_lo + 16) * 32 + pos * 8) =
      make_uint2(packed[2], packed[3]);
}

__global__ void pack_w_kernel(const int* __restrict__ qweight,
                              const int* __restrict__ qzeros,
                              const float* __restrict__ scales,
                              int8_t* __restrict__ packedB) {
  const int g    = blockIdx.x * blockDim.x + threadIdx.x;
  const int lane = g & 31;
  const int nt   = (g >> 5) & (NTILES - 1);
  const int kt   = g >> 13;

  const int n   = nt * 16 + (lane & 15);
  const int grp = kt >> 1;
  const int zw  = qzeros[grp * (OUT_F / 8) + (n >> 3)];
  const int8_t z = (int8_t)(((zw >> (4 * (n & 7))) & 15) << 4);
  const float  s = scales[grp * OUT_F + n];

  uint32_t dw[8];
#pragma unroll
  for (int d = 0; d < 8; ++d) {
    const int klocal = (d >> 1) * 16 + ((lane >= 16) ? 8 : 0) + (d & 1) * 4;
    const int kg = kt * 64 + klocal;
    const int w  = qweight[(size_t)(kg >> 3) * OUT_F + n];
    const int j  = kg & 7;
    uint32_t o = 0;
#pragma unroll
    for (int b = 0; b < 4; ++b) {
      const int8_t wv  = (int8_t)(((w >> (4 * (j + b))) & 15) << 4);
      const int   diff = (int)(int8_t)(wv - z);
      const int   q    = q8clamp(s * (float)diff);
      o |= (uint32_t)(q & 255) << (8 * b);
    }
    dw[d] = o;
  }
  uint4* dst = (uint4*)(packedB + (((size_t)kt * NTILES + nt) * 32 + lane) * 32);
  dst[0] = make_uint4(dw[0], dw[1], dw[2], dw[3]);
  dst[1] = make_uint4(dw[4], dw[5], dw[6], dw[7]);
}

__global__ void gemm_iu8_kernel(const int8_t* __restrict__ packedA,
                                const float* __restrict__ xscale,
                                const int8_t* __restrict__ packedB,
                                const float* __restrict__ int8_scales,
                                const float* __restrict__ bias,
                                float* __restrict__ out) {
  const int lane = threadIdx.x & 31;
  const int wave = threadIdx.x >> 5;
  const int mtile0 = blockIdx.y * 16 + wave * 2;
  const int nt0    = blockIdx.x * 4;

  const int8_t* aBase = packedA + (((size_t)mtile0 * KTILES) * 32 + lane) * 32;
  const int8_t* bBase = packedB + (((size_t)nt0) * 32 + lane) * 32;
  const size_t aMStride = (size_t)KTILES * 32 * 32;
  const size_t bNStride = (size_t)32 * 32;
  const size_t bKStride = (size_t)NTILES * 32 * 32;

  v8i acc0[4] = {};
  v8i acc1[4] = {};

#pragma unroll 1
  for (int kt = 0; kt < KTILES; ++kt) {
    const v8i A0 = load_frag(aBase + (size_t)kt * 1024);
    const v8i A1 = load_frag(aBase + (size_t)kt * 1024 + aMStride);
    const int8_t* bk = bBase + (size_t)kt * bKStride;
#pragma unroll
    for (int t = 0; t < 4; ++t) {
      const v8i B = load_frag(bk + (size_t)t * bNStride);
      acc0[t] = __builtin_amdgcn_wmma_i32_16x16x64_iu8(
          true, A0, true, B, acc0[t], false, false);
      acc1[t] = __builtin_amdgcn_wmma_i32_16x16x64_iu8(
          true, A1, true, B, acc1[t], false, false);
    }
  }

  const int colBase = nt0 * 16 + (lane & 15);
  float ds[4], bs[4];
#pragma unroll
  for (int t = 0; t < 4; ++t) {
    ds[t] = int8_scales[colBase + t * 16];
    bs[t] = bias[colBase + t * 16];
  }

#pragma unroll
  for (int f = 0; f < 2; ++f) {
    const v8i* acc = f ? acc1 : acc0;
    const int row0 = (mtile0 + f) * 16 + ((lane >= 16) ? 8 : 0);
    float xs[8];
#pragma unroll
    for (int v = 0; v < 8; ++v) xs[v] = xscale[row0 + v];
#pragma unroll
    for (int t = 0; t < 4; ++t) {
      const int col = colBase + t * 16;
#pragma unroll
      for (int v = 0; v < 8; ++v) {
        const float o = (float)acc[t][v] * (xs[v] * ds[t]) + bs[t];
        out[(size_t)(row0 + v) * OUT_F + col] = o;
      }
    }
  }
}

extern "C" void kernel_launch(void* const* d_in, const int* in_sizes, int n_in,
                              void* d_out, int out_size, void* d_ws, size_t ws_size,
                              hipStream_t stream) {
  const float* x           = (const float*)d_in[0];
  const int*   qweight     = (const int*)d_in[1];
  const int*   qzeros      = (const int*)d_in[2];
  const float* scales      = (const float*)d_in[3];
  const float* int8_scales = (const float*)d_in[4];
  const float* bias        = (const float*)d_in[5];

  const int Ntok = in_sizes[0] / IN_F;

  uint8_t* ws = (uint8_t*)d_ws;
  size_t off = 0;
  int8_t* packedA = (int8_t*)(ws + off);
  off += (size_t)Ntok * IN_F;
  off = (off + 255) & ~(size_t)255;
  float* xs = (float*)(ws + off);
  off += (size_t)Ntok * sizeof(float);
  off = (off + 255) & ~(size_t)255;
  int8_t* packedB = (int8_t*)(ws + off);
  off += (size_t)KTILES * NTILES * 32 * 32;

  quant_act_kernel<<<Ntok, 256, 0, stream>>>(x, packedA, xs);

  const int packThreads = KTILES * NTILES * 32;
  pack_w_kernel<<<packThreads / 256, 256, 0, stream>>>(qweight, qzeros, scales, packedB);

  dim3 grid(OUT_F / 64, Ntok / 256);
  gemm_iu8_kernel<<<grid, 256, 0, stream>>>(packedA, xs, packedB, int8_scales, bias,
                                            (float*)d_out);
}
